// GreenFunction_83923660964245
// MI455X (gfx1250) — matrix operand order repaired and checked by reading, not compiled, not run on hardware
//
#include <hip/hip_runtime.h>

// ---------------------------------------------------------------------------
// Problem constants (from the reference)
// ---------------------------------------------------------------------------
#define D_   2
#define B_   64
#define P_   512
#define NV_  64
#define H_   64
#define TW_  128
#define NQ_  (B_ * NV_ + B_)   // 4160 transport rows (batched + single)
#define NTILE_ (NQ_ / 16)      // 260 row-tiles of 16 (exact)

typedef __attribute__((ext_vector_type(16))) _Float16 v16h;
typedef __attribute__((ext_vector_type(8)))  float    v8f;

// D = A(16x32 f16) x B(32x16 f16) + C(16x16 f32)
__device__ inline v8f wmma_f16(v16h a, v16h b, v8f c) {
  return __builtin_amdgcn_wmma_f32_16x16x32_f16(
      /*neg_a=*/false, a, /*neg_b=*/false, b,
      /*c_mod=*/(short)0, c, /*reuse_a=*/false, /*reuse_b=*/false);
}

// A fragment: 16x32 f16 tile from row-major [16][ldk] source (LDS).
// ISA layout (wave32): lane = {M = lane&15, half = lane>>4};
// element e -> K = (e<8 ? e : e+8) + 8*half   (+32*ks for the K-step)
__device__ inline v16h load_a_frag(const _Float16* src, int ldk, int lane, int ks) {
  const int M = lane & 15, hf = lane >> 4;
  v16h a;
#pragma unroll
  for (int e = 0; e < 16; ++e) {
    const int K = ((e < 8) ? e : (e + 8)) + 8 * hf + 32 * ks;
    a[e] = src[M * ldk + K];
  }
  return a;
}

// B fragment: 32x16 f16 tile from row-major [K][N] weight matrix (global f16).
// ISA layout (wave32): lane = column N = (lane&15)+16*nt; element e -> K = (e<8 ? e : e+8) + 8*half.
__device__ inline v16h load_b_frag(const _Float16* __restrict__ W, int N,
                                   int lane, int ks, int nt) {
  const int col = (lane & 15) + 16 * nt, hf = lane >> 4;
  v16h b;
#pragma unroll
  for (int e = 0; e < 16; ++e)
    b[e] = W[(size_t)(32 * ks + ((e < 8) ? e : (e + 8)) + 8 * hf) * N + col];
  return b;
}

// ---------------------------------------------------------------------------
// Kernel 0: one-time f32 -> f16 conversion of the static GEMM weights
// ---------------------------------------------------------------------------
__global__ void k_convert(const float* __restrict__ aW1,
                          const float* __restrict__ tW1,
                          const float* __restrict__ tW2,
                          _Float16* __restrict__ aW1h,
                          _Float16* __restrict__ tW1h,
                          _Float16* __restrict__ tW2h) {
  int i = blockIdx.x * blockDim.x + threadIdx.x;
  if (i < H_ * H_)   aW1h[i] = (_Float16)aW1[i];
  if (i < TW_ * TW_) { tW1h[i] = (_Float16)tW1[i]; tW2h[i] = (_Float16)tW2[i]; }
}

// ---------------------------------------------------------------------------
// Kernel 1: _coeff attention.  One block (4 waves) per query; each wave owns
// 128 of the P=512 positions as 8 tiles of 16 rows.  Layer2 (64x64) is WMMA.
// ---------------------------------------------------------------------------
__global__ __launch_bounds__(128) void k_coeff(
    const float* __restrict__ coords, const float* __restrict__ vel,
    const float* __restrict__ pos,    const float* __restrict__ sigma,
    const float* __restrict__ aW0,    const float* __restrict__ ab0,
    const _Float16* __restrict__ aW1h,const float* __restrict__ ab1,
    const float* __restrict__ aW2,    const float* __restrict__ ab2,
    float* __restrict__ coeff)
{
  __shared__ _Float16 h1s[4][16][H_];     // per-wave layer-1 activations (A src)
  __shared__ float pls[4][16];            // per-wave pl (mask) values
  __shared__ float slog[P_];              // masked logits for softmax
  __shared__ float saW0[6 * H_];
  __shared__ float sab0[H_], sab1[H_], saW2[H_];
  __shared__ float red[128];

  const int q    = blockIdx.x;
  const int tid  = threadIdx.x, wave = tid >> 5, lane = tid & 31;
  const int M    = lane & 15,   hf   = lane >> 4;

  // query features x, v
  float x0, x1, v0, v1;
  if (q < B_ * NV_) {
    const int b = q >> 6, j = q & 63;
    x0 = coords[4 * b];     x1 = coords[4 * b + 1];
    v0 = vel[2 * j];        v1 = vel[2 * j + 1];
  } else {
    const int b = q - B_ * NV_;
    x0 = coords[4 * b];     x1 = coords[4 * b + 1];
    v0 = coords[4 * b + 2]; v1 = coords[4 * b + 3];
  }
  const float inv = rsqrtf(v0 * v0 + v1 * v1 + 1e-16f);
  const float g0 = v0 * inv, g1 = v1 * inv;      // ang_g
  const float ab2v = ab2[0];

  for (int i = tid; i < 6 * H_; i += 128) saW0[i] = aW0[i];
  if (tid < H_) { sab0[tid] = ab0[tid]; sab1[tid] = ab1[tid]; saW2[tid] = aW2[tid]; }
  __syncthreads();

  // Preload all B fragments of aW1 (64x64): 2 K-steps x 4 N-tiles, reused 8x.
  v16h bf[2][4];
#pragma unroll
  for (int ks = 0; ks < 2; ++ks)
#pragma unroll
    for (int nt = 0; nt < 4; ++nt)
      bf[ks][nt] = load_b_frag(aW1h, H_, lane, ks, nt);

  for (int t = 0; t < 8; ++t) {
    const int p0 = wave * 128 + t * 16;

    // ---- layer 1 (K=6, scalar): lane handles row M, columns hf*32..hf*32+31
    {
      const int p = p0 + M;
      const float r0 = x0 - pos[2 * p], r1 = x1 - pos[2 * p + 1];
      const float rd = sqrtf(r0 * r0 + r1 * r1 + 1e-16f);
      const float pl = r0 * g0 + r1 * g1;
      const float al = pl / (rd + 1e-8f);
      if (hf == 0) pls[wave][M] = pl;
      const float f0 = x0, f1 = x1, f2 = v0, f3 = v1, f4 = al, f5 = pl;
#pragma unroll 8
      for (int nn = 0; nn < 32; ++nn) {
        const int n = hf * 32 + nn;
        float acc = sab0[n];
        acc = fmaf(f0, saW0[0 * H_ + n], acc);
        acc = fmaf(f1, saW0[1 * H_ + n], acc);
        acc = fmaf(f2, saW0[2 * H_ + n], acc);
        acc = fmaf(f3, saW0[3 * H_ + n], acc);
        acc = fmaf(f4, saW0[4 * H_ + n], acc);
        acc = fmaf(f5, saW0[5 * H_ + n], acc);
        h1s[wave][M][n] = (_Float16)fmaxf(acc, 0.f);
      }
    }
    __syncthreads();   // uniform: every wave runs exactly 8 iterations

    // ---- layer 2 via WMMA (K=64 -> 2 x 16x16x32), layer 3 folded in epilogue
    const v16h a0 = load_a_frag(&h1s[wave][0][0], H_, lane, 0);
    const v16h a1 = load_a_frag(&h1s[wave][0][0], H_, lane, 1);
    float part[8];
#pragma unroll
    for (int r = 0; r < 8; ++r) part[r] = 0.f;
#pragma unroll
    for (int nt = 0; nt < 4; ++nt) {
      v8f c = {};
      c = wmma_f16(a0, bf[0][nt], c);
      c = wmma_f16(a1, bf[1][nt], c);
      const int   n  = 16 * nt + M;
      const float b1 = sab1[n], w2 = saW2[n];
#pragma unroll
      for (int r = 0; r < 8; ++r)            // relu(+ab1) then * aW2
        part[r] = fmaf(fmaxf(c[r] + b1, 0.f), w2, part[r]);
    }
    // reduce the 16 N-lanes of each half-wave
#pragma unroll
    for (int m2 = 1; m2 < 16; m2 <<= 1)
#pragma unroll
      for (int r = 0; r < 8; ++r) part[r] += __shfl_xor(part[r], m2, 32);
    if (M == 0) {
#pragma unroll
      for (int r = 0; r < 8; ++r) {
        const int mm = r + 8 * hf;           // C layout: M = r + 8*half
        slog[p0 + mm] = (pls[wave][mm] > 0.f) ? (part[r] + ab2v) : -1e30f;
      }
    }
  }
  __syncthreads();

  // ---- masked softmax over P, attn = sum(attw * sigma), coeff = exp(-attn)
  float mx = -INFINITY;
#pragma unroll
  for (int i = 0; i < 4; ++i) mx = fmaxf(mx, slog[tid + 128 * i]);
  red[tid] = mx; __syncthreads();
  for (int s = 64; s > 0; s >>= 1) { if (tid < s) red[tid] = fmaxf(red[tid], red[tid + s]); __syncthreads(); }
  const float gmax = red[0]; __syncthreads();

  float se = 0.f, sa = 0.f;
#pragma unroll
  for (int i = 0; i < 4; ++i) {
    const int p = tid + 128 * i;
    const float e = expf(slog[p] - gmax);
    se += e; sa = fmaf(e, sigma[p], sa);
  }
  red[tid] = se; __syncthreads();
  for (int s = 64; s > 0; s >>= 1) { if (tid < s) red[tid] += red[tid + s]; __syncthreads(); }
  const float tote = red[0]; __syncthreads();
  red[tid] = sa; __syncthreads();
  for (int s = 64; s > 0; s >>= 1) { if (tid < s) red[tid] += red[tid + s]; __syncthreads(); }
  if (tid == 0) coeff[q] = expf(-(red[0] / tote));
}

// ---------------------------------------------------------------------------
// Kernel 2: transport MLP (9 -> 128 -> 128 -> 128, exp).  One 16-row tile
// per wave; layers 2 & 3 are 128-K WMMA (4 K-steps x 8 N-tiles each).
// ---------------------------------------------------------------------------
__global__ __launch_bounds__(128) void k_transport(
    const float* __restrict__ coords, const float* __restrict__ cp,
    const float* __restrict__ vel,    const float* __restrict__ coeff,
    const float* __restrict__ tW0,    const float* __restrict__ tb0,
    const _Float16* __restrict__ tW1h,const float* __restrict__ tb1,
    const _Float16* __restrict__ tW2h,const float* __restrict__ tb2,
    float* __restrict__ gmat)
{
  __shared__ _Float16 hbuf[4][16][TW_];   // per-wave activations (A source)
  __shared__ float stW0[9 * TW_];
  __shared__ float stb0[TW_], stb1[TW_], stb2[TW_];

  const int tid  = threadIdx.x, wave = tid >> 5, lane = tid & 31;
  const int M    = lane & 15,   hf   = lane >> 4;
  const int tile = blockIdx.x * 4 + wave;
  const int row0 = tile * 16;

  for (int i = tid; i < 9 * TW_; i += 128) stW0[i] = tW0[i];
  if (tid < TW_) { stb0[tid] = tb0[tid]; stb1[tid] = tb1[tid]; stb2[tid] = tb2[tid]; }
  __builtin_prefetch(tW1h, 0, 1);         // -> global_prefetch_b8 (L2-warm the f16 weights)
  __builtin_prefetch(tW2h, 0, 1);
  __syncthreads();

  // features of this lane's row: inp = [x, v, xp, vp, coeff]
  const int q = row0 + M;
  float f[9];
  int b;
  if (q < B_ * NV_) {
    b = q >> 6; const int j = q & 63;
    f[2] = vel[2 * j]; f[3] = vel[2 * j + 1];
  } else {
    b = q - B_ * NV_;
    f[2] = coords[4 * b + 2]; f[3] = coords[4 * b + 3];
  }
  f[0] = coords[4 * b]; f[1] = coords[4 * b + 1];
  f[4] = cp[4 * b];     f[5] = cp[4 * b + 1];
  f[6] = cp[4 * b + 2]; f[7] = cp[4 * b + 3];
  f[8] = coeff[q];

  // ---- layer 1 (K=9, scalar): lane handles row M, columns hf*64..hf*64+63
#pragma unroll 4
  for (int nn = 0; nn < 64; ++nn) {
    const int n = hf * 64 + nn;
    float acc = stb0[n];
#pragma unroll
    for (int k = 0; k < 9; ++k) acc = fmaf(f[k], stW0[k * TW_ + n], acc);
    hbuf[wave][M][n] = (_Float16)fmaxf(acc, 0.f);
  }
  __syncthreads();

  // ---- layer 2 (128x128 WMMA), write relu back into hbuf
  v16h A[4];
#pragma unroll
  for (int ks = 0; ks < 4; ++ks) A[ks] = load_a_frag(&hbuf[wave][0][0], TW_, lane, ks);
#pragma unroll
  for (int nt = 0; nt < 8; ++nt) {
    v8f c = {};
#pragma unroll
    for (int ks = 0; ks < 4; ++ks) c = wmma_f16(A[ks], load_b_frag(tW1h, TW_, lane, ks, nt), c);
    const int n = 16 * nt + M;
    const float bb = stb1[n];
#pragma unroll
    for (int r = 0; r < 8; ++r)              // C layout: row = r + 8*half, col = n
      hbuf[wave][r + 8 * hf][n] = (_Float16)fmaxf(c[r] + bb, 0.f);
  }
  __syncthreads();

  // ---- layer 3 (128x128 WMMA) + exp, store to gmat
#pragma unroll
  for (int ks = 0; ks < 4; ++ks) A[ks] = load_a_frag(&hbuf[wave][0][0], TW_, lane, ks);
#pragma unroll
  for (int nt = 0; nt < 8; ++nt) {
    v8f c = {};
#pragma unroll
    for (int ks = 0; ks < 4; ++ks) c = wmma_f16(A[ks], load_b_frag(tW2h, TW_, lane, ks, nt), c);
    const int n = 16 * nt + M;
    const float bb = stb2[n];
#pragma unroll
    for (int r = 0; r < 8; ++r)
      gmat[(size_t)(row0 + r + 8 * hf) * TW_ + n] = expf(fmaxf(c[r] + bb, 0.f));
  }
}

// ---------------------------------------------------------------------------
// Kernel 3: scattering residual loop + final projection (<2% of FLOPs).
// One block per b; thread tid owns output column w = tid; cur columns live
// in registers (64 VGPRs), r_mid staged through LDS.
// ---------------------------------------------------------------------------
__global__ __launch_bounds__(128) void k_scatter(
    const float* __restrict__ gmat, const float* __restrict__ sk,
    const float* __restrict__ vw,   const float* __restrict__ selfk,
    const float* __restrict__ rW,   const float* __restrict__ rb,
    const float* __restrict__ fW,   const float* __restrict__ fb,
    float* __restrict__ out)
{
  __shared__ float rbuf[NV_][TW_];       // r_mid[i][w]
  __shared__ float sres[NV_ * NV_];      // (1 - selfK) * vw
  __shared__ float swj[NV_];             // (1 - sk[b]) * vw
  __shared__ float red[128];

  const int b = blockIdx.x, w = threadIdx.x;
  for (int i = w; i < NV_ * NV_; i += 128) sres[i] = (1.f - selfk[i]) * vw[i & 63];
  if (w < NV_) swj[w] = (1.f - sk[b * NV_ + w]) * vw[w];
  __syncthreads();

  float curr[NV_];
#pragma unroll 8
  for (int j = 0; j < NV_; ++j) curr[j] = gmat[(size_t)(b * NV_ + j) * TW_ + w];

  float contrib = 0.f;                   // outs[0] contribution
#pragma unroll 8
  for (int j = 0; j < NV_; ++j) contrib = fmaf(swj[j], curr[j], contrib);

  for (int l = 0; l < 2; ++l) {
    // r_mid[i][w] = sum_j res_w[i][j] * cur[j][w]
#pragma unroll 1
    for (int i = 0; i < NV_; ++i) {
      float acc = 0.f;
#pragma unroll 8
      for (int j = 0; j < NV_; ++j) acc = fmaf(sres[i * NV_ + j], curr[j], acc);
      rbuf[i][w] = acc;
    }
    __syncthreads();
    // r2[i][w] = relu(sum_u r_mid[i][u] * rW[l][u][w] + rb[l][w]); cur += r2
    float acc2[NV_];
#pragma unroll 8
    for (int i = 0; i < NV_; ++i) acc2[i] = 0.f;
    const float* rwl = rW + (size_t)l * TW_ * TW_;
#pragma unroll 1
    for (int u = 0; u < TW_; ++u) {
      const float wv = rwl[(size_t)u * TW_ + w];
#pragma unroll 8
      for (int i = 0; i < NV_; ++i) acc2[i] = fmaf(rbuf[i][u], wv, acc2[i]);
    }
    const float bias = rb[l * TW_ + w];
#pragma unroll 8
    for (int i = 0; i < NV_; ++i) {
      curr[i] += fmaxf(acc2[i] + bias, 0.f);
      contrib = fmaf(swj[i], curr[i], contrib);   // outs[l+1] contribution
    }
    __syncthreads();
  }

  const float gb = gmat[(size_t)(B_ * NV_ + b) * TW_ + w];  // single-query g
  red[w] = (gb + contrib) * fW[w];
  __syncthreads();
  for (int s = 64; s > 0; s >>= 1) { if (w < s) red[w] += red[w + s]; __syncthreads(); }
  if (w == 0) out[b] = red[0] + fb[0];
}

// ---------------------------------------------------------------------------
// Launch
// ---------------------------------------------------------------------------
extern "C" void kernel_launch(void* const* d_in, const int* in_sizes, int n_in,
                              void* d_out, int out_size, void* d_ws, size_t ws_size,
                              hipStream_t stream) {
  (void)in_sizes; (void)n_in; (void)out_size; (void)ws_size;
  const float* coords = (const float*)d_in[0];
  const float* cp     = (const float*)d_in[1];
  const float* sk     = (const float*)d_in[2];
  const float* pos    = (const float*)d_in[3];
  const float* sigma  = (const float*)d_in[4];
  const float* vel    = (const float*)d_in[5];
  const float* vw     = (const float*)d_in[6];
  const float* selfk  = (const float*)d_in[7];
  const float* aW0 = (const float*)d_in[8];
  const float* ab0 = (const float*)d_in[9];
  const float* aW1 = (const float*)d_in[10];
  const float* ab1 = (const float*)d_in[11];
  const float* aW2 = (const float*)d_in[12];
  const float* ab2 = (const float*)d_in[13];
  const float* tW0 = (const float*)d_in[14];
  const float* tb0 = (const float*)d_in[15];
  const float* tW1 = (const float*)d_in[16];
  const float* tb1 = (const float*)d_in[17];
  const float* tW2 = (const float*)d_in[18];
  const float* tb2 = (const float*)d_in[19];
  const float* rW  = (const float*)d_in[20];
  const float* rb  = (const float*)d_in[21];
  const float* fW  = (const float*)d_in[22];
  const float* fb  = (const float*)d_in[23];

  // Workspace layout (total ~2.2 MB)
  char* ws = (char*)d_ws;
  _Float16* aW1h = (_Float16*)(ws + 0);        //  8 KB
  _Float16* tW1h = (_Float16*)(ws + 8192);     // 32 KB
  _Float16* tW2h = (_Float16*)(ws + 40960);    // 32 KB
  float*    coeff = (float*)(ws + 73728);      // NQ_ * 4 = 16640 B
  float*    gmat  = (float*)(ws + 90368);      // NQ_ * 128 * 4 = 2,129,920 B

  k_convert  <<<64,   256, 0, stream>>>(aW1, tW1, tW2, aW1h, tW1h, tW2h);
  k_coeff    <<<NQ_,  128, 0, stream>>>(coords, vel, pos, sigma,
                                        aW0, ab0, aW1h, ab1, aW2, ab2, coeff);
  k_transport<<<NTILE_ / 4, 128, 0, stream>>>(coords, cp, vel, coeff,
                                              tW0, tb0, tW1h, tb1, tW2h, tb2, gmat);
  k_scatter  <<<B_,   128, 0, stream>>>(gmat, sk, vw, selfk, rW, rb, fW, fb,
                                        (float*)d_out);
}
